// Model_31885837205925
// MI455X (gfx1250) — compile-verified, matrix operand order checked by reading, not run on hardware
//
#include <hip/hip_runtime.h>
#include <math.h>

#define Bb 8
#define Ll 4096
#define Dd 768
#define Tt 768
#define Aa 32
#define Kk 8
#define Hh 384
#define LmM 4095
#define G3 1152   // 3*H
#define NG 2304   // 2*3*H (both directions)
#define NR 33     // A + 1 (all_reps rows)

typedef __bf16 bf16_t;
typedef bf16_t v16bf __attribute__((ext_vector_type(16)));
typedef float  v8f   __attribute__((ext_vector_type(8)));

// Load a 16x32 (bf16) WMMA A/B fragment slice for this lane from an f32 row.
// Layout per CDNA5 ISA 7.12.2: lane half=0 holds K=[kk0..kk0+7] in elems 0-7 and
// K=[kk0+16..kk0+23] in elems 8-15; half=1 holds K=+8 / K=+24 runs.
__device__ inline v16bf ld_ab_f32(const float* row, int kk0, int half) {
  const float* p0 = row + kk0 + half * 8;
  const float* p1 = row + kk0 + 16 + half * 8;
  v16bf r;
#pragma unroll
  for (int i = 0; i < 8; ++i) {
    r[i]     = (bf16_t)p0[i];
    r[8 + i] = (bf16_t)p1[i];
  }
  return r;
}

__device__ inline v8f wmma_bf16(v16bf a, v16bf b, v8f c) {
  return __builtin_amdgcn_wmma_f32_16x16x32_bf16(false, a, false, b, (short)0, c,
                                                 false, false);
}

__device__ inline float sigf(float x) { return 1.0f / (1.0f + expf(-x)); }

// ---------------------------------------------------------------------------
// Kernel 0: nvalid[b] = sum(enc_attn[b,1:]); zero ctxsum accumulator.
__global__ void init_kernel(const int* __restrict__ enc, float* __restrict__ nvalid,
                            float* __restrict__ ctxsum) {
  __shared__ int sred[256];
  int b = blockIdx.x, tid = threadIdx.x;
  for (int d = tid; d < Dd; d += 256) ctxsum[b * Dd + d] = 0.0f;
  int s = 0;
  for (int l = 1 + tid; l < Ll; l += 256) s += enc[b * Ll + l];
  sred[tid] = s;
  __syncthreads();
  for (int off = 128; off > 0; off >>= 1) {
    if (tid < off) sred[tid] += sred[tid + off];
    __syncthreads();
  }
  if (tid == 0) nvalid[b] = (float)sred[0];
}

// ---------------------------------------------------------------------------
// Kernel 1: rep_ac[b,a,:] = mean of tokens over [start,end] inclusive.
__global__ void span_kernel(const float* __restrict__ hidden,
                            const int* __restrict__ starts,
                            const int* __restrict__ ends,
                            float* __restrict__ repac) {
  int blk = blockIdx.x;
  int b = blk >> 5;
  int s = starts[blk], e = ends[blk];
  float inv = 1.0f / (float)(e - s + 1);
  int tid = threadIdx.x;
  float a0 = 0.f, a1 = 0.f, a2 = 0.f;
  for (int l = s; l <= e; ++l) {
    const float* tok = hidden + ((size_t)b * Ll + 1 + l) * Dd;
    a0 += tok[tid];
    a1 += tok[tid + 256];
    a2 += tok[tid + 512];
  }
  float* dst = repac + (size_t)blk * Dd;
  dst[tid] = a0 * inv;
  dst[tid + 256] = a1 * inv;
  dst[tid + 512] = a2 * inv;
}

// ---------------------------------------------------------------------------
// Kernel 2: ctxsum[b,:] += sum over l < nvalid-1 of tokens (chunked, atomics).
__global__ void ctxsum_kernel(const float* __restrict__ hidden,
                              const float* __restrict__ nvalid,
                              float* __restrict__ ctxsum) {
  int b = blockIdx.y, c = blockIdx.x;
  int limit = (int)nvalid[b] - 1;
  if (limit > LmM) limit = LmM;
  int l0 = c * 64, l1 = l0 + 64;
  if (l1 > limit) l1 = limit;
  if (l0 >= l1) return;
  int tid = threadIdx.x;
  float a0 = 0.f, a1 = 0.f, a2 = 0.f;
  for (int l = l0; l < l1; ++l) {
    const float* tok = hidden + ((size_t)b * Ll + 1 + l) * Dd;
    a0 += tok[tid];
    a1 += tok[tid + 256];
    a2 += tok[tid + 512];
  }
  atomicAdd(&ctxsum[b * Dd + tid], a0);
  atomicAdd(&ctxsum[b * Dd + tid + 256], a1);
  atomicAdd(&ctxsum[b * Dd + tid + 512], a2);
}

// ---------------------------------------------------------------------------
// Kernel 3: GI[256,2304] = rep_ac @ concat(wih_f, wih_b)^T  (no bias here).
// One 16x16 output tile per wave; K swept in steps of 32 via bf16 WMMA.
__global__ void gi_gemm(const float* __restrict__ repac,
                        const float* __restrict__ wf,
                        const float* __restrict__ wb, float* __restrict__ gi) {
  int wid = blockIdx.x * 8 + (threadIdx.x >> 5);  // 0..2303
  int lane = threadIdx.x & 31;
  int lm = lane & 15, half = lane >> 4;
  int mtile = wid / 144, ntile = wid % 144;
  int n = ntile * 16 + lm;
  const float* wrow = (n < G3) ? (wf + (size_t)n * Dd) : (wb + (size_t)(n - G3) * Dd);
  const float* arow = repac + (size_t)(mtile * 16 + lm) * Dd;
  v8f acc = {};
  for (int kk0 = 0; kk0 < Dd; kk0 += 32) {
    v16bf av = ld_ab_f32(arow, kk0, half);
    v16bf bv = ld_ab_f32(wrow, kk0, half);
    acc = wmma_bf16(av, bv, acc);
  }
#pragma unroll
  for (int v = 0; v < 8; ++v) {
    int m = mtile * 16 + half * 8 + v;
    gi[(size_t)m * NG + n] = acc[v];
  }
}

// ---------------------------------------------------------------------------
// Kernel 4: ctx[8,768] = (ctxsum / (nvalid-1)) @ ctx_w^T + ctx_b   (M padded to 16)
__global__ void ctx_gemm(const float* __restrict__ ctxsum,
                         const float* __restrict__ nvalid,
                         const float* __restrict__ ctx_w,
                         const float* __restrict__ ctx_b,
                         float* __restrict__ ctxout) {
  int wid = blockIdx.x * 8 + (threadIdx.x >> 5);  // 0..47
  int lane = threadIdx.x & 31;
  int lm = lane & 15, half = lane >> 4;
  int n = wid * 16 + lm;
  const float* wrow = ctx_w + (size_t)n * Dd;
  float scale = (lm < Bb) ? (1.0f / (nvalid[lm] - 1.0f)) : 0.0f;
  const float* arow = ctxsum + (size_t)((lm < Bb) ? lm : 0) * Dd;
  v8f acc = {};
  for (int kk0 = 0; kk0 < Dd; kk0 += 32) {
    const float* p0 = arow + kk0 + half * 8;
    const float* p1 = arow + kk0 + 16 + half * 8;
    v16bf av;
#pragma unroll
    for (int i = 0; i < 8; ++i) {
      av[i] = (bf16_t)(p0[i] * scale);
      av[8 + i] = (bf16_t)(p1[i] * scale);
    }
    v16bf bv = ld_ab_f32(wrow, kk0, half);
    acc = wmma_bf16(av, bv, acc);
  }
  float bias = ctx_b[n];
#pragma unroll
  for (int v = 0; v < 8; ++v) {
    int m = half * 8 + v;
    if (m < Bb) ctxout[(size_t)m * Tt + n] = acc[v] + bias;
  }
}

// ---------------------------------------------------------------------------
// Kernel 5: bi-GRU recurrence. One block per direction (sequential scan).
// Per step: GH[16,1152] = h @ whh^T via WMMA (8 waves x 9 tiles), then gates.
// Dynamic LDS: GH (16*1152 f32) + h (16*384 f32) = 98304 B (< 320 KB/WGP).
__global__ void gru_kernel(const float* __restrict__ gi,
                           const float* __restrict__ whh_f,
                           const float* __restrict__ whh_b,
                           const float* __restrict__ bih_f,
                           const float* __restrict__ bhh_f,
                           const float* __restrict__ bih_b,
                           const float* __restrict__ bhh_b,
                           float* __restrict__ gruout) {
  extern __shared__ float smem[];
  float* GH = smem;              // 16*1152
  float* hb = smem + 16 * G3;    // 16*384 (rows 8..15 stay zero = M padding)
  int dir = blockIdx.x;
  const float* whh = dir ? whh_b : whh_f;
  const float* bih = dir ? bih_b : bih_f;
  const float* bhh = dir ? bhh_b : bhh_f;
  int tid = threadIdx.x;
  for (int i = tid; i < 16 * Hh; i += 256) hb[i] = 0.0f;
  __syncthreads();
  int w = tid >> 5, lane = tid & 31, lm = lane & 15, half = lane >> 4;
  for (int t = 0; t < Aa; ++t) {
    int step = dir ? (Aa - 1 - t) : t;
    // phase 1: GH = h @ whh^T (WMMA)
    for (int tt = 0; tt < 9; ++tt) {
      int n0 = (w * 9 + tt) * 16;
      const float* arow = hb + lm * Hh;
      const float* brow = whh + (size_t)(n0 + lm) * Hh;
      v8f acc = {};
      for (int kk0 = 0; kk0 < Hh; kk0 += 32) {
        v16bf av = ld_ab_f32(arow, kk0, half);
        v16bf bv = ld_ab_f32(brow, kk0, half);
        acc = wmma_bf16(av, bv, acc);
      }
#pragma unroll
      for (int v = 0; v < 8; ++v) {
        int m = half * 8 + v;
        GH[m * G3 + n0 + lm] = acc[v];
      }
    }
    __syncthreads();
    // phase 2: gate math + state update (rows 0..7 = real batches)
    for (int idx = tid; idx < Bb * Hh; idx += 256) {
      int b = idx / Hh, j = idx - b * Hh;
      const float* g = gi + (size_t)(b * Aa + step) * NG + dir * G3;
      float ir = g[j] + bih[j];
      float iz = g[Hh + j] + bih[Hh + j];
      float inn = g[2 * Hh + j] + bih[2 * Hh + j];
      float hr = GH[b * G3 + j] + bhh[j];
      float hz = GH[b * G3 + Hh + j] + bhh[Hh + j];
      float hn = GH[b * G3 + 2 * Hh + j] + bhh[2 * Hh + j];
      float r = sigf(ir + hr);
      float z = sigf(iz + hz);
      float nn = tanhf(inn + r * hn);
      float hp = hb[b * Hh + j];
      float h2 = (1.0f - z) * nn + z * hp;
      hb[b * Hh + j] = h2;
      gruout[(size_t)(b * Aa + step) * Tt + dir * Hh + j] = h2;
    }
    __syncthreads();
  }
}

// ---------------------------------------------------------------------------
// Kernel 6: q[b,a] = gru_out . wq ; kk[b,r] = all_reps . wk  (r=0..32)
__global__ void qk_kernel(const float* __restrict__ gruout,
                          const float* __restrict__ ctxout,
                          const float* __restrict__ wfindm,
                          float* __restrict__ qv, float* __restrict__ kkv) {
  __shared__ float sq[256], sk[256];
  int r = blockIdx.x, b = blockIdx.y, tid = threadIdx.x;
  const float* vec =
      (r < Aa) ? (gruout + (size_t)(b * Aa + r) * Tt) : (ctxout + (size_t)b * Tt);
  const float* wq = wfindm;
  const float* wk = wfindm + Tt;
  float dq = 0.f, dk = 0.f;
  for (int d = tid; d < Tt; d += 256) {
    float x = vec[d];
    dq += x * wq[d];
    dk += x * wk[d];
  }
  sq[tid] = dq;
  sk[tid] = dk;
  __syncthreads();
  for (int off = 128; off > 0; off >>= 1) {
    if (tid < off) { sq[tid] += sq[tid + off]; sk[tid] += sk[tid + off]; }
    __syncthreads();
  }
  if (tid == 0) {
    if (r < Aa) qv[b * Aa + r] = sq[0];
    kkv[b * NR + r] = sk[0];
  }
}

// ---------------------------------------------------------------------------
// Kernel 7: top-8 per (b,a). sigmoid is monotone and q is constant across j,
// so ranking == ranking of kk[b,j]; strict '>' gives stable lowest-index ties.
__global__ void topk_kernel(const float* __restrict__ qv,
                            const float* __restrict__ kkv,
                            float* __restrict__ vals, float* __restrict__ indf) {
  int t = threadIdx.x;
  int b = t >> 5, a = t & 31;
  float q = qv[b * Aa + a];
  const float* kk = kkv + b * NR;
  unsigned long long chosen = 0ull;
  for (int sel = 0; sel < Kk; ++sel) {
    float best = -INFINITY;
    int bi = 0;
    for (int j = 0; j < NR; ++j) {
      if (chosen & (1ull << j)) continue;
      float v = kk[j];
      if (v > best) { best = v; bi = j; }
    }
    chosen |= (1ull << bi);
    vals[(b * Aa + a) * Kk + sel] = sigf(q + best);
    indf[(b * Aa + a) * Kk + sel] = (float)bi;
  }
}

// ---------------------------------------------------------------------------
// Kernel 8: v1[b,a,s,:] = all_reps[b, ind[b,a,s], :]
__global__ void gather_kernel(const float* __restrict__ gruout,
                              const float* __restrict__ ctxout,
                              const float* __restrict__ indf,
                              float* __restrict__ v1) {
  int blk = blockIdx.x;  // (b*32 + a)*8 + s, 0..2047
  int idx = (int)indf[blk];
  int b = blk >> 8;  // 256 entries per batch
  const float* src = (idx < Aa) ? (gruout + (size_t)(b * Aa + idx) * Tt)
                                : (ctxout + (size_t)b * Tt);
  float* dst = v1 + (size_t)blk * Tt;
  int tid = threadIdx.x;
  dst[tid] = src[tid];
  dst[tid + 256] = src[tid + 256];
  dst[tid + 512] = src[tid + 512];
}

// ---------------------------------------------------------------------------
extern "C" void kernel_launch(void* const* d_in, const int* in_sizes, int n_in,
                              void* d_out, int out_size, void* d_ws, size_t ws_size,
                              hipStream_t stream) {
  const float* hidden = (const float*)d_in[0];
  const int* enc = (const int*)d_in[1];
  const int* sst = (const int*)d_in[2];
  const int* sen = (const int*)d_in[3];
  const float* wih_f = (const float*)d_in[4];
  const float* whh_f = (const float*)d_in[5];
  const float* bih_f = (const float*)d_in[6];
  const float* bhh_f = (const float*)d_in[7];
  const float* wih_b = (const float*)d_in[8];
  const float* whh_b = (const float*)d_in[9];
  const float* bih_b = (const float*)d_in[10];
  const float* bhh_b = (const float*)d_in[11];
  const float* ctx_w = (const float*)d_in[12];
  const float* ctx_b = (const float*)d_in[13];
  const float* wfindm = (const float*)d_in[14];

  float* out = (float*)d_out;
  float* gruout = out;               // 8*32*768      = 196608
  float* ctxout = out + 196608;      // 8*768         = 6144
  float* v1 = out + 202752;          // 8*32*8*768    = 1572864
  float* vals = out + 1775616;       // 8*32*8        = 2048
  float* indf = out + 1777664;       // 8*32*8        = 2048

  float* ws = (float*)d_ws;
  float* nvalid = ws;                // 8
  float* ctxsum = ws + 16;           // 6144
  float* repac = ws + 6208;          // 256*768  = 196608
  float* gi = ws + 202816;           // 256*2304 = 589824
  float* qv = ws + 792640;           // 256
  float* kkv = ws + 792896;          // 264

  init_kernel<<<Bb, 256, 0, stream>>>(enc, nvalid, ctxsum);
  span_kernel<<<Bb * Aa, 256, 0, stream>>>(hidden, sst, sen, repac);
  ctxsum_kernel<<<dim3(64, Bb), 256, 0, stream>>>(hidden, nvalid, ctxsum);
  gi_gemm<<<288, 256, 0, stream>>>(repac, wih_f, wih_b, gi);
  ctx_gemm<<<6, 256, 0, stream>>>(ctxsum, nvalid, ctx_w, ctx_b, ctxout);
  gru_kernel<<<2, 256, (16 * G3 + 16 * Hh) * sizeof(float), stream>>>(
      gi, whh_f, whh_b, bih_f, bhh_f, bih_b, bhh_b, gruout);
  qk_kernel<<<dim3(NR, Bb), 256, 0, stream>>>(gruout, ctxout, wfindm, qv, kkv);
  topk_kernel<<<1, 256, 0, stream>>>(qv, kkv, vals, indf);
  gather_kernel<<<Bb * Aa * Kk, 256, 0, stream>>>(gruout, ctxout, indf, v1);
}
